// Mamba2Block_73126113182308
// MI455X (gfx1250) — compile-verified, matrix operand order checked by reading, not run on hardware
//
#include <hip/hip_runtime.h>
#include <hip/hip_bf16.h>
#include <cmath>

// ---------------------------------------------------------------------------
// Types for CDNA5 WMMA (wave32)
// ---------------------------------------------------------------------------
typedef __attribute__((ext_vector_type(16))) __bf16 v16bf;
typedef __attribute__((ext_vector_type(8)))  __bf16 v8bf;
typedef __attribute__((ext_vector_type(8)))  float  v8f;

// ---------------------------------------------------------------------------
// f32 -> bf16 conversion (weights / activations)
// ---------------------------------------------------------------------------
__global__ __launch_bounds__(256) void f32_to_bf16_kernel(
    const float* __restrict__ in, __bf16* __restrict__ out, long n)
{
    long i = (long)blockIdx.x * blockDim.x + threadIdx.x;
    long stride = (long)gridDim.x * blockDim.x;
    for (; i < n; i += stride) out[i] = (__bf16)in[i];
}

// ---------------------------------------------------------------------------
// RMSNorm: per-row (Dm) normalize, scale by norm_w, emit bf16
// ---------------------------------------------------------------------------
__global__ __launch_bounds__(256) void rmsnorm_bf16_kernel(
    const float* __restrict__ x, const float* __restrict__ w,
    __bf16* __restrict__ out, int Dm)
{
    const long row = blockIdx.x;
    const float* xr = x + row * (long)Dm;
    float ss = 0.f;
    for (int i = threadIdx.x; i < Dm; i += 256) { float v = xr[i]; ss += v * v; }
    __shared__ float red[256];
    red[threadIdx.x] = ss;
    __syncthreads();
    for (int off = 128; off > 0; off >>= 1) {
        if ((int)threadIdx.x < off) red[threadIdx.x] += red[threadIdx.x + off];
        __syncthreads();
    }
    float scale = rsqrtf(red[0] / (float)Dm + 1e-6f);
    for (int i = threadIdx.x; i < Dm; i += 256)
        out[row * (long)Dm + i] = (__bf16)(xr[i] * scale * w[i]);
}

// ---------------------------------------------------------------------------
// WMMA GEMM:  C[M,N] = A[M,K](bf16) * W[N,K](bf16)^T   (row-major, K%32==0)
// One wave computes a 16x64 tile (A-fragment reused across 4 B-fragments).
// 8 waves per block -> block covers 128 rows x 64 cols.
// epi: 0 = plain, 1 = softplus(v + bias[col]), 2 = v + addmat[row,col]
// Optional bf16 mirror of the output for chained GEMM consumption.
//
// Fragment layout per CDNA5 ISA (16-bit A 16x32, wave32):
//   lane&15 -> M (or N for B-operand), lane>>4 selects K-half (kb = 0 or 8),
//   each lane holds K = kb+0..7 and kb+16..23  => two contiguous 16B loads.
// ---------------------------------------------------------------------------
__device__ inline v16bf load_frag_row(const __bf16* p)
{
    v8bf lo = *(const v8bf*)(p);       // k = kb + 0..7
    v8bf hi = *(const v8bf*)(p + 16);  // k = kb + 16..23
    return __builtin_shufflevector(lo, hi, 0, 1, 2, 3, 4, 5, 6, 7,
                                           8, 9, 10, 11, 12, 13, 14, 15);
}

#define GEMM_NT 4

__global__ __launch_bounds__(256) void gemm_bf16_wmma_kernel(
    const __bf16* __restrict__ A, const __bf16* __restrict__ W,
    float* __restrict__ C, __bf16* __restrict__ Cbf,
    const float* __restrict__ bias, const float* __restrict__ addmat,
    int M, int N, int K, int epi)
{
    const int lane = threadIdx.x & 31;
    const int wave = threadIdx.x >> 5;
    const int m0 = (blockIdx.y * 8 + wave) * 16;
    const int n0 = blockIdx.x * (16 * GEMM_NT);
    if (m0 >= M) return;                 // wave-uniform: EXEC stays all-ones

    const int r  = lane & 15;            // M row (A) / N col (B)
    const int kb = (lane >> 4) << 3;     // K-half base: 0 or 8

    const __bf16* arow = A + (long)(m0 + r) * K + kb;
    const __bf16* wrow[GEMM_NT];
#pragma unroll
    for (int t = 0; t < GEMM_NT; ++t)
        wrow[t] = W + (long)(n0 + t * 16 + r) * K + kb;

    v8f acc[GEMM_NT] = {};

    for (int k0 = 0; k0 < K; k0 += 32) {
        v16bf af = load_frag_row(arow + k0);
        __builtin_prefetch(arow + k0 + 64, 0, 1);   // global_prefetch_b8
#pragma unroll
        for (int t = 0; t < GEMM_NT; ++t) {
            v16bf bfrag = load_frag_row(wrow[t] + k0);
            acc[t] = __builtin_amdgcn_wmma_f32_16x16x32_bf16(
                false, af, false, bfrag, (short)0, acc[t], false, false);
        }
    }

    // C/D layout: VGPR rr holds M = m0 + rr + (lane>=16 ? 8 : 0), N = n-tile + (lane&15)
    const int crow = m0 + ((lane >> 4) << 3);
#pragma unroll
    for (int t = 0; t < GEMM_NT; ++t) {
        const int col = n0 + t * 16 + r;
#pragma unroll
        for (int rr = 0; rr < 8; ++rr) {
            long off = (long)(crow + rr) * N + col;
            float v = acc[t][rr];
            if (epi == 1) {
                v += bias[col];
                v = (v > 20.f) ? v : log1pf(expf(v));     // softplus
            } else if (epi == 2) {
                v += addmat[off];                          // residual add
            }
            if (C)   C[off]   = v;
            if (Cbf) Cbf[off] = (__bf16)v;
        }
    }
}

// ---------------------------------------------------------------------------
// Causal depthwise conv1d (D_CONV=4) + bias + SiLU on x_in half of xz.
// Emits f32 (for scan / gate) and bf16 (for projection GEMMs).
// ---------------------------------------------------------------------------
__global__ __launch_bounds__(256) void conv_silu_kernel(
    const float* __restrict__ xz, const float* __restrict__ cw,
    const float* __restrict__ cb, float* __restrict__ xc,
    __bf16* __restrict__ xcbf, int S, int Di, long total)
{
    long idx = (long)blockIdx.x * blockDim.x + threadIdx.x;
    long stride = (long)gridDim.x * blockDim.x;
    const long D2 = 2L * Di;
    for (; idx < total; idx += stride) {
        long bs = idx / Di;
        int  d  = (int)(idx - bs * Di);
        int  s  = (int)(bs % S);
        float acc = cb[d];
#pragma unroll
        for (int j = 0; j < 4; ++j) {
            int back = 3 - j;                  // tap j reads x[s - (3-j)]
            if (s >= back)
                acc += cw[d * 4 + j] * xz[(bs - back) * D2 + d];
        }
        float v = acc / (1.f + expf(-acc));    // SiLU
        xc[idx]   = v;
        xcbf[idx] = (__bf16)v;
    }
}

// ---------------------------------------------------------------------------
// Selective scan, software-pipelined.
//  - 8 lanes per channel, 8 states each held in VGPRs (SPLIT=8)
//    => 256 blocks of 256 threads: more independent recurrence chains
//       co-resident per WGP to hide each other's latencies.
//  - Double-buffered LDS for the per-step B/C vectors: step s+1's global
//    loads (x, dt, B, C) are issued BEFORE step s's math, so load latency
//    overlaps the recurrence instead of serializing with it.
//  - One __syncthreads per step (publish next buffer).
//  - Partial C.h reduced across the 8 lanes with __shfl_xor (wave32).
// Block = 256 threads = 32 channels; Di*8/256 blocks per batch => same b.
// ---------------------------------------------------------------------------
__global__ __launch_bounds__(256) void scan_kernel(
    const float* __restrict__ xc, const float* __restrict__ dt,
    const float* __restrict__ Bm, const float* __restrict__ Cm,
    const float* __restrict__ A_log, float* __restrict__ y,
    int S, int Di)
{
    constexpr int DS = 64, SPLIT = 8, NS = DS / SPLIT;   // 8 states/lane
    const int  tid  = threadIdx.x;
    const long g    = (long)blockIdx.x * blockDim.x + tid;
    const int  part = (int)(g & (SPLIT - 1));
    const long c    = g >> 3;                            // channel index
    const int  d    = (int)(c % Di);
    const int  b    = (int)(c / Di);
    const int  nb   = part * NS;

    float h[NS], a[NS];
#pragma unroll
    for (int i = 0; i < NS; ++i) {
        h[i] = 0.f;
        a[i] = -expf(A_log[(long)d * DS + nb + i]);
    }

    __shared__ float sB[2][DS], sC[2][DS];
    const long bS = (long)b * S;

    // ---- prologue: stage step 0 ----
    {
        long rowBC0 = bS * DS;
        if (tid < DS)          sB[0][tid]      = Bm[rowBC0 + tid];
        else if (tid < 2 * DS) sC[0][tid - DS] = Cm[rowBC0 + tid - DS];
    }
    long  rowX = bS * Di + d;
    float xt   = xc[rowX];
    float dtt  = dt[rowX];
    __syncthreads();

    int p = 0;
    for (int s = 0; s < S; ++s) {
        // ---- issue step s+1 loads early (off the dependency chain) ----
        float xt_n = 0.f, dtt_n = 0.f, bc_n = 0.f;
        if (s + 1 < S) {
            long rowXn = rowX + Di;
            xt_n  = xc[rowXn];
            dtt_n = dt[rowXn];
            long rowBCn = (bS + s + 1) * DS;
            if (tid < DS)          bc_n = Bm[rowBCn + tid];
            else if (tid < 2 * DS) bc_n = Cm[rowBCn + tid - DS];
        }

        // ---- step s recurrence ----
        float acc = 0.f;
#pragma unroll
        for (int i = 0; i < NS; ++i) {
            h[i] += dtt * (a[i] * h[i] + sB[p][nb + i] * xt);
            acc  += sC[p][nb + i] * h[i];
        }
        acc += __shfl_xor(acc, 1, 32);
        acc += __shfl_xor(acc, 2, 32);
        acc += __shfl_xor(acc, 4, 32);
        if (part == 0) y[rowX] = acc;

        // ---- publish step s+1 buffer ----
        if (s + 1 < S) {
            if (tid < DS)          sB[p ^ 1][tid]      = bc_n;
            else if (tid < 2 * DS) sC[p ^ 1][tid - DS] = bc_n;
        }
        __syncthreads();
        xt = xt_n; dtt = dtt_n; p ^= 1; rowX += Di;
    }
}

// ---------------------------------------------------------------------------
// Gate: y = (y_scan + D[d]*xc) * silu(z), z from second half of xz. -> bf16
// ---------------------------------------------------------------------------
__global__ __launch_bounds__(256) void gate_kernel(
    const float* __restrict__ yscan, const float* __restrict__ xc,
    const float* __restrict__ xz, const float* __restrict__ Dp,
    __bf16* __restrict__ ybf, int Di, long total)
{
    long idx = (long)blockIdx.x * blockDim.x + threadIdx.x;
    long stride = (long)gridDim.x * blockDim.x;
    const long D2 = 2L * Di;
    for (; idx < total; idx += stride) {
        long bs = idx / Di;
        int  d  = (int)(idx - bs * Di);
        float z = xz[bs * D2 + Di + d];
        float v = (yscan[idx] + Dp[d] * xc[idx]) * (z / (1.f + expf(-z)));
        ybf[idx] = (__bf16)v;
    }
}

// ---------------------------------------------------------------------------
// Host launcher
// ---------------------------------------------------------------------------
static inline char* ws_take(char*& cur, size_t bytes)
{
    char* p = cur;
    cur += (bytes + 255) & ~(size_t)255;
    return p;
}

extern "C" void kernel_launch(void* const* d_in, const int* in_sizes, int n_in,
                              void* d_out, int out_size, void* d_ws, size_t ws_size,
                              hipStream_t stream)
{
    const float* x      = (const float*)d_in[0];
    const float* norm_w = (const float*)d_in[1];
    const float* in_w   = (const float*)d_in[2];
    const float* conv_w = (const float*)d_in[3];
    const float* conv_b = (const float*)d_in[4];
    const float* A_log  = (const float*)d_in[5];
    const float* D_p    = (const float*)d_in[6];
    const float* dt_w   = (const float*)d_in[7];
    const float* dt_b   = (const float*)d_in[8];
    const float* xp_w   = (const float*)d_in[9];
    const float* Bp_w   = (const float*)d_in[10];
    const float* Cp_w   = (const float*)d_in[11];
    const float* out_w  = (const float*)d_in[12];

    const long Dm = in_sizes[1];            // 1024
    const long Di = in_sizes[4];            // 2048 (conv_b)
    const long Ds = in_sizes[5] / Di;       // 64   (A_log)
    const long Dr = in_sizes[7] / Di;       // 64   (dt_w)
    const long BS = in_sizes[0] / Dm;       // B*S = 8192
    const long Bb = 4;                      // batch (reference fixes B=4)
    const long S  = BS / Bb;                // 2048
    (void)n_in; (void)out_size; (void)ws_size;

    // ---- workspace carve-out -------------------------------------------
    char* cur = (char*)d_ws;
    __bf16* xn_bf    = (__bf16*)ws_take(cur, BS * Dm * 2);
    __bf16* w_in_bf  = (__bf16*)ws_take(cur, 2 * Di * Dm * 2);
    __bf16* w_xp_bf  = (__bf16*)ws_take(cur, Dr * Di * 2);
    __bf16* w_dt_bf  = (__bf16*)ws_take(cur, Di * Dr * 2);
    __bf16* w_Bp_bf  = (__bf16*)ws_take(cur, Ds * Di * 2);
    __bf16* w_Cp_bf  = (__bf16*)ws_take(cur, Ds * Di * 2);
    __bf16* w_out_bf = (__bf16*)ws_take(cur, Dm * Di * 2);
    float*  xz       = (float*)ws_take(cur, BS * 2 * Di * 4);
    float*  xc       = (float*)ws_take(cur, BS * Di * 4);
    __bf16* xc_bf    = (__bf16*)ws_take(cur, BS * Di * 2);
    float*  tmp_dt   = (float*)ws_take(cur, BS * Dr * 4);
    __bf16* tmp_dtbf = (__bf16*)ws_take(cur, BS * Dr * 2);
    float*  dt_full  = (float*)ws_take(cur, BS * Di * 4);
    float*  Bm       = (float*)ws_take(cur, BS * Ds * 4);
    float*  Cm       = (float*)ws_take(cur, BS * Ds * 4);
    float*  yscan    = (float*)ws_take(cur, BS * Di * 4);
    __bf16* y_bf     = (__bf16*)ws_take(cur, BS * Di * 2);

    auto cvt = [&](const float* s, __bf16* dpt, long n) {
        long g = (n + 255) / 256; if (g > 16384) g = 16384;
        f32_to_bf16_kernel<<<dim3((unsigned)g), dim3(256), 0, stream>>>(s, dpt, n);
    };
    auto gemm = [&](const __bf16* A, const __bf16* W, float* C, __bf16* Cbf,
                    const float* bias, const float* add,
                    long M, long N, long K, int epi) {
        dim3 grid((unsigned)(N / 64), (unsigned)(M / 128));
        gemm_bf16_wmma_kernel<<<grid, dim3(256), 0, stream>>>(
            A, W, C, Cbf, bias, add, (int)M, (int)N, (int)K, epi);
    };

    // 1) bf16 weight mirrors
    cvt(in_w,  w_in_bf,  2 * Di * Dm);
    cvt(xp_w,  w_xp_bf,  Dr * Di);
    cvt(dt_w,  w_dt_bf,  Di * Dr);
    cvt(Bp_w,  w_Bp_bf,  Ds * Di);
    cvt(Cp_w,  w_Cp_bf,  Ds * Di);
    cvt(out_w, w_out_bf, Dm * Di);

    // 2) RMSNorm -> bf16
    rmsnorm_bf16_kernel<<<dim3((unsigned)BS), dim3(256), 0, stream>>>(
        x, norm_w, xn_bf, (int)Dm);

    // 3) in-projection: xz[BS, 2*Di] = xn @ in_w^T
    gemm(xn_bf, w_in_bf, xz, nullptr, nullptr, nullptr, BS, 2 * Di, Dm, 0);

    // 4) causal depthwise conv + SiLU
    {
        long total = BS * Di;
        long g = (total + 255) / 256; if (g > 8192) g = 8192;
        conv_silu_kernel<<<dim3((unsigned)g), dim3(256), 0, stream>>>(
            xz, conv_w, conv_b, xc, xc_bf, (int)S, (int)Di, total);
    }

    // 5) dt stage 1: tmp[BS, Dr] = xc @ xp_w^T  (also bf16 mirror)
    gemm(xc_bf, w_xp_bf, tmp_dt, tmp_dtbf, nullptr, nullptr, BS, Dr, Di, 0);

    // 6) dt stage 2: dt[BS, Di] = softplus(tmp @ dt_w^T + dt_b)
    gemm(tmp_dtbf, w_dt_bf, dt_full, nullptr, dt_b, nullptr, BS, Di, Dr, 1);

    // 7) B / C projections: [BS, Ds]
    gemm(xc_bf, w_Bp_bf, Bm, nullptr, nullptr, nullptr, BS, Ds, Di, 0);
    gemm(xc_bf, w_Cp_bf, Cm, nullptr, nullptr, nullptr, BS, Ds, Di, 0);

    // 8) selective scan (software-pipelined, double-buffered LDS)
    {
        unsigned g = (unsigned)(Bb * Di * 8 / 256);   // 256 blocks
        scan_kernel<<<dim3(g), dim3(256), 0, stream>>>(
            xc, dt_full, Bm, Cm, A_log, yscan, (int)S, (int)Di);
    }

    // 9) gate: y = (yscan + D*xc) * silu(z) -> bf16
    {
        long total = BS * Di;
        long g = (total + 255) / 256; if (g > 8192) g = 8192;
        gate_kernel<<<dim3((unsigned)g), dim3(256), 0, stream>>>(
            yscan, xc, xz, D_p, y_bf, (int)Di, total);
    }

    // 10) out-projection + residual: out = x + y @ out_w^T
    gemm(y_bf, w_out_bf, (float*)d_out, nullptr, nullptr, x, BS, Dm, Di, 2);
}
